// TrainablePQC_23115513987350
// MI455X (gfx1250) — matrix operand order repaired and checked by reading, not compiled, not run on hardware
//
#include <hip/hip_runtime.h>

// ---------------------------------------------------------------------------
// TrainablePQC on MI455X (gfx1250, wave32, WMMA).
//
// out[b,q] = sum_s sign(q,s) * |(U psi_b)_s|^2
//   psi_b  : real product state from RY encoding  -> [4096,1024] f16
//   U      : variational unitary (shared)         -> Ur/Ui f16, pre-swizzled
// Two f16 WMMA GEMMs (Re/Im) + analytic sign reduction.
//
// Round-1 refinement: cap K-loop unrolling at 4 so the GEMM wave stays under
// 256 VGPRs (round-0 build fully unrolled 32 K-steps, spilled into the
// VGPR-MSB space -> s_set_vgpr_msb between every WMMA and ~2 waves/SIMD
// occupancy). 4-deep b128 load pipelining still covers L2 latency.
// ---------------------------------------------------------------------------

typedef __attribute__((ext_vector_type(16))) _Float16 v16h;
typedef __attribute__((ext_vector_type(8)))  _Float16 v8h;
typedef __attribute__((ext_vector_type(8)))  float    v8f;

#define NQ      10
#define DIM     1024          // 2^NQ
#define NLAYERS 4
#define BATCHSZ 4096
#define PSI_STRIDE 1032       // halves; 2064B row stride -> bank-conflict-free b128 DS loads

// ---------------------------------------------------------------------------
// Kernel 1: build U^T as pre-swizzled WMMA B-fragments (f16 Re / Im planes).
// One workgroup per basis column t; 1024-amplitude complex state kept in LDS.
// Fragment layout (16-bit B 32x16, contiguous K order as in A): element (k,n) of tile
// (sTile,kTile) lives at halves[ ((sTile*32+kTile)*512) + lane*16 + (k&15) ],
// lane = ((k>>4)<<4) | n.  GEMM kernel then loads each lane's v16h contiguously.
// ---------------------------------------------------------------------------
__global__ __launch_bounds__(256)
void pqc_build_unitary(const float* __restrict__ params,
                       _Float16* __restrict__ fragR,
                       _Float16* __restrict__ fragI)
{
    __shared__ __align__(16) float2 st[DIM];
    const int t   = blockIdx.x;
    const int tid = threadIdx.x;

    for (int s = tid; s < DIM; s += 256)
        st[s] = make_float2(s == t ? 1.f : 0.f, 0.f);
    __syncthreads();

    for (int l = 0; l < NLAYERS; ++l) {
        // fused RZ(theta_z) @ RY(theta_y) on each qubit
        for (int q = 0; q < NQ; ++q) {
            const float hy = 0.5f * params[l * 2 * NQ + q];
            const float hz = 0.5f * params[l * 2 * NQ + NQ + q];
            const float cy = __cosf(hy), sy = __sinf(hy);
            const float cz = __cosf(hz), sz = __sinf(hz);
            // row0 * e^{-i hz}, row1 * e^{+i hz}; ry = [[cy,-sy],[sy,cy]]
            const float g00r =  cz*cy, g00i = -sz*cy;
            const float g01r = -cz*sy, g01i =  sz*sy;
            const float g10r =  cz*sy, g10i =  sz*sy;
            const float g11r =  cz*cy, g11i =  sz*cy;
            const int p  = NQ - 1 - q;      // bit position (qubit 0 = MSB)
            const int pm = 1 << p;
            for (int j = tid; j < DIM / 2; j += 256) {
                const int lo = j & (pm - 1);
                const int hi = (j >> p) << (p + 1);
                const int i0 = hi | lo;
                const int i1 = i0 | pm;
                const float2 a0 = st[i0], a1 = st[i1];
                float2 r0, r1;
                r0.x = g00r*a0.x - g00i*a0.y + g01r*a1.x - g01i*a1.y;
                r0.y = g00r*a0.y + g00i*a0.x + g01r*a1.y + g01i*a1.x;
                r1.x = g10r*a0.x - g10i*a0.y + g11r*a1.x - g11i*a1.y;
                r1.y = g10r*a0.y + g10i*a0.x + g11r*a1.y + g11i*a1.x;
                st[i0] = r0; st[i1] = r1;
            }
            __syncthreads();
        }
        // CNOT ring: q -> (q+1)%NQ, applied sequentially
        for (int q = 0; q < NQ; ++q) {
            const int pc = NQ - 1 - q;
            const int pt = NQ - 1 - ((q + 1) % NQ);
            for (int s = tid; s < DIM; s += 256) {
                if (((s >> pc) & 1) && !((s >> pt) & 1)) {
                    const int s2 = s | (1 << pt);
                    const float2 tmp = st[s];
                    st[s]  = st[s2];
                    st[s2] = tmp;
                }
            }
            __syncthreads();
        }
    }

    // st[s] = U[s, t].  Scatter B[t,s] = U[s,t] into fragment layout.
    const int k      = t & 31;
    const int kTile  = t >> 5;
    const int laneHi = (k >> 4) << 4;
    const int jh     = k & 15;
    for (int s = tid; s < DIM; s += 256) {
        const int sTile = s >> 4;
        const int lane  = laneHi | (s & 15);
        const size_t off = (size_t)(sTile * 32 + kTile) * 512 + lane * 16 + jh;
        const float2 a = st[s];
        fragR[off] = (_Float16)a.x;
        fragI[off] = (_Float16)a.y;
    }
}

// ---------------------------------------------------------------------------
// Kernel 2: per-WG 16 batch rows. psi tile in LDS, WMMA GEMM over all 64
// N-tiles x 32 K-steps, squared-magnitude + analytic Z-sign contraction.
// ---------------------------------------------------------------------------
__global__ __launch_bounds__(256)
void pqc_expval_wmma(const float* __restrict__ x,
                     const _Float16* __restrict__ fragR,
                     const _Float16* __restrict__ fragI,
                     float* __restrict__ out)
{
    __shared__ __align__(32) _Float16 psi[16 * PSI_STRIDE];
    __shared__ float accLDS[16 * NQ];

    const int tid = threadIdx.x;
    const int b0  = blockIdx.x * 16;

    if (tid < 16 * NQ) accLDS[tid] = 0.f;

    // psi[m][t] = prod_q (bit_q(t) ? sin : cos)(x[b0+m,q]/2); 16 threads/row.
    {
        const int m   = tid >> 4;
        const int sub = tid & 15;
        float c[NQ], s[NQ];
        #pragma unroll
        for (int q = 0; q < NQ; ++q) {
            const float h = 0.5f * x[(size_t)(b0 + m) * NQ + q];
            c[q] = __cosf(h);
            s[q] = __sinf(h);
        }
        for (int t = sub; t < DIM; t += 16) {
            float v = 1.f;
            #pragma unroll
            for (int q = 0; q < NQ; ++q)
                v *= ((t >> (NQ - 1 - q)) & 1) ? s[q] : c[q];
            psi[m * PSI_STRIDE + t] = (_Float16)v;
        }
    }
    __syncthreads();

    const int lane = tid & 31;
    const int wave = tid >> 5;
    const int n    = lane & 15;   // WMMA C/D column index of this lane
    const int grp  = lane >> 4;   // 0: rows 0-7 / K 0-15 ; 1: rows 8-15 / K 16-31

    // per-lane output accumulators: 8 M-rows x 10 qubits
    float acc[8][NQ];
    #pragma unroll
    for (int i = 0; i < 8; ++i)
        #pragma unroll
        for (int q = 0; q < NQ; ++q) acc[i][q] = 0.f;

    const _Float16* psiRow = &psi[n * PSI_STRIDE];   // A-matrix row M = lane&15

    #pragma unroll 1                                  // keep VGPR pressure flat
    for (int sT = wave; sT < 64; sT += 8) {           // each wave: 8 N-tiles
        v8f ampR = {};
        v8f ampI = {};
        const _Float16* bR = fragR + (size_t)sT * 32 * 512 + lane * 16;
        const _Float16* bI = fragI + (size_t)sT * 32 * 512 + lane * 16;
        #pragma unroll 4                              // 4-deep load pipelining,
        for (int kT = 0; kT < 32; ++kT) {             // wave stays < 256 VGPRs
            const int k0 = kT * 32;
            // A fragment (contiguous K order k = 16*grp + e, matching B): two aligned b128 DS loads
            const v8h alo = *(const v8h*)(psiRow + k0 + grp * 16);
            const v8h ahi = *(const v8h*)(psiRow + k0 + grp * 16 + 8);
            v16h a;
            #pragma unroll
            for (int e = 0; e < 8; ++e) { a[e] = alo[e]; a[8 + e] = ahi[e]; }
            // B fragments: pre-swizzled, each lane's 16 halves contiguous (32B)
            const v16h br = *(const v16h*)(bR + (size_t)kT * 512);
            const v16h bi = *(const v16h*)(bI + (size_t)kT * 512);
            ampR = __builtin_amdgcn_wmma_f32_16x16x32_f16(
                       false, a, false, br, (short)0, ampR, false, false);
            ampI = __builtin_amdgcn_wmma_f32_16x16x32_f16(
                       false, a, false, bi, (short)0, ampI, false, false);
        }
        // sign(q, s=sT*16+n): bits 9..4 come from sT (tile-uniform),
        // bits 3..0 come from n (lane-constant).
        float sHi[6];
        #pragma unroll
        for (int q = 0; q < 6; ++q)
            sHi[q] = ((sT >> (5 - q)) & 1) ? -1.f : 1.f;
        float sLo[4];
        #pragma unroll
        for (int q = 0; q < 4; ++q)
            sLo[q] = ((n >> (3 - q)) & 1) ? -1.f : 1.f;
        #pragma unroll
        for (int i = 0; i < 8; ++i) {
            const float p = ampR[i] * ampR[i] + ampI[i] * ampI[i];
            #pragma unroll
            for (int q = 0; q < 6; ++q) acc[i][q]     += p * sHi[q];
            #pragma unroll
            for (int q = 0; q < 4; ++q) acc[i][6 + q] += p * sLo[q];
        }
    }

    // reduce over the 16 N-lanes of each half-wave (xor masks stay in-group)
    #pragma unroll
    for (int i = 0; i < 8; ++i)
        #pragma unroll
        for (int q = 0; q < NQ; ++q) {
            float v = acc[i][q];
            v += __shfl_xor(v, 1);
            v += __shfl_xor(v, 2);
            v += __shfl_xor(v, 4);
            v += __shfl_xor(v, 8);
            acc[i][q] = v;
        }

    // combine across the 8 waves (ds_add_f32), then store 16x10 outputs
    if (n == 0) {
        const int mbase = grp * 8;   // VGPR i holds M=i (grp0) / M=8+i (grp1)
        #pragma unroll
        for (int i = 0; i < 8; ++i)
            #pragma unroll
            for (int q = 0; q < NQ; ++q)
                atomicAdd(&accLDS[(mbase + i) * NQ + q], acc[i][q]);
    }
    __syncthreads();
    if (tid < 16 * NQ)
        out[(size_t)b0 * NQ + tid] = accLDS[tid];
}

// ---------------------------------------------------------------------------
extern "C" void kernel_launch(void* const* d_in, const int* in_sizes, int n_in,
                              void* d_out, int out_size, void* d_ws, size_t ws_size,
                              hipStream_t stream) {
    const float* x      = (const float*)d_in[0];   // [4096,10] f32
    const float* params = (const float*)d_in[1];   // [4,20]    f32
    float*       out    = (float*)d_out;           // [4096,10] f32

    _Float16* fragR = (_Float16*)d_ws;                       // 2 MB
    _Float16* fragI = fragR + (size_t)DIM * DIM;             // 2 MB

    pqc_build_unitary<<<DIM, 256, 0, stream>>>(params, fragR, fragI);
    pqc_expval_wmma<<<BATCHSZ / 16, 256, 0, stream>>>(x, fragR, fragI, out);
}
